// SpikingTransformer3D_52862457479927
// MI455X (gfx1250) — matrix operand order repaired and checked by reading, not compiled, not run on hardware
//
#include <hip/hip_runtime.h>

// ---------------- CDNA5 WMMA types / wrapper ----------------
typedef __attribute__((ext_vector_type(16))) _Float16 v16h;
typedef __attribute__((ext_vector_type(8)))  float    v8f;

__device__ __forceinline__ v8f wmma_f16(v16h a, v16h b, v8f c) {
  // D = A(16x32 f16) * B(32x16 f16) + C(16x16 f32)
  return __builtin_amdgcn_wmma_f32_16x16x32_f16(false, a, false, b, (short)0, c,
                                                false, false);
}

// ---------------- model constants ----------------
#define TAU_INV   0.5f                   // 1/tau, tau=2
#define VTH       1.0f
#define BN_SCALE  0.9999950000374996f    // 1/sqrt(1+1e-5)
#define LN_EPSF   1e-5f
#define NTOK      65536                  // B * 32^3 tokens after patch embed
#define NWIN      1024                   // B * 8^3 windows
#define DEC_POS   524288                 // B * 64^3 decoder voxels

// ---------------- WMMA fragment loaders (contiguous k = 16h + s order, wave32) ----------------
// A: 16x32 f16 row-major, lanes0-15: K 0..15, lanes16-31: K 16..31
__device__ __forceinline__ v16h frag_a_f16(const _Float16* __restrict__ A, int lda) {
  const int lane = threadIdx.x & 31, m = lane & 15, h = lane >> 4;
  v16h a;
#pragma unroll
  for (int i = 0; i < 8; ++i) {
    const int k0 = 2 * i + 16 * h;
    a[2 * i]     = A[m * lda + k0];
    a[2 * i + 1] = A[m * lda + k0 + 1];
  }
  return a;
}

__device__ __forceinline__ v16h frag_a_f16_klim(const _Float16* __restrict__ A, int lda, int klim) {
  const int lane = threadIdx.x & 31, m = lane & 15, h = lane >> 4;
  v16h a;
#pragma unroll
  for (int i = 0; i < 8; ++i) {
    const int k0 = 2 * i + 16 * h;
    a[2 * i]     = (k0     < klim) ? A[m * lda + k0]     : (_Float16)0.0f;
    a[2 * i + 1] = (k0 + 1 < klim) ? A[m * lda + k0 + 1] : (_Float16)0.0f;
  }
  return a;
}

// B: 32x16 f16 [k][n] from LDS, lanes0-15: K 0..15, lanes16-31: K 16..31
__device__ __forceinline__ v16h frag_b_f16(const _Float16* __restrict__ B, int ldb) {
  const int lane = threadIdx.x & 31, n = lane & 15, h = lane >> 4;
  v16h b;
#pragma unroll
  for (int i = 0; i < 8; ++i) {
    const int k = 2 * i + 16 * h;
    b[2 * i]     = B[k * ldb + n];
    b[2 * i + 1] = B[(k + 1) * ldb + n];
  }
  return b;
}

// B fragment = transpose of an LDS row-major f16 tile: B[k][n] = S[n*ld + k], zero-padded past klim
__device__ __forceinline__ v16h frag_b_f16T_klim(const _Float16* __restrict__ S, int ld, int klim) {
  const int lane = threadIdx.x & 31, n = lane & 15, h = lane >> 4;
  v16h b;
#pragma unroll
  for (int i = 0; i < 8; ++i) {
    const int k = 2 * i + 16 * h;
    b[2 * i]     = (k     < klim) ? S[n * ld + k]     : (_Float16)0.0f;
    b[2 * i + 1] = (k + 1 < klim) ? S[n * ld + k + 1] : (_Float16)0.0f;
  }
  return b;
}

// B fragment from pre-packed weight: one aligned 32B load per lane.
// Pack layout: [(kt*ntn + nt)][lane][16 halves]
__device__ __forceinline__ v16h frag_b_pack(const _Float16* __restrict__ P,
                                            int ntn, int kt, int nt) {
  const int lane = threadIdx.x & 31;
  const _Float16* p = P + (((size_t)(kt * ntn + nt) * 32) + lane) * 16;
  return *(const v16h*)p;
}

// LIF: v += (x-v)/tau; spike = v>=1; hard reset
__device__ __forceinline__ float lif_fire(float* __restrict__ vmem, float x) {
  float v = *vmem;
  v += (x - v) * TAU_INV;
  const float s = (v >= VTH) ? 1.0f : 0.0f;
  *vmem = v * (1.0f - s);
  return s;
}

// token g (window-major) -> (b, dz,dy,dx) on the 32^3 grid
__device__ __forceinline__ void tok_decode(int g, int& b, int& dz, int& dy, int& dx) {
  b = g >> 15;
  const int r = g & 32767;
  const int wb = r >> 6, tl = r & 63;
  dz = ((wb >> 6) << 2)       + (tl >> 4);
  dy = (((wb >> 3) & 7) << 2) + ((tl >> 2) & 3);
  dx = ((wb & 7) << 2)        + (tl & 3);
}

// ---------------- zero init ----------------
__global__ void k_zero(float* __restrict__ p, long long n) {
  long long i = (long long)blockIdx.x * blockDim.x + threadIdx.x;
  const long long st = (long long)gridDim.x * blockDim.x;
  for (; i < n; i += st) p[i] = 0.0f;
}

// ---------------- weight packers: f32 -> WMMA-B-fragment f16 ----------------
// From torch Linear layout W[n][k] (ld=K).
__global__ void k_pack_wT(const float* __restrict__ W, int K, int N,
                          _Float16* __restrict__ out) {
  const int lid = blockIdx.x * blockDim.x + threadIdx.x;
  const int ntn = N >> 4, ntk = K >> 5;
  if (lid >= ntk * ntn * 32) return;
  const int lane = lid & 31, tile = lid >> 5;
  const int nt = tile % ntn, kt = tile / ntn;
  const int n = lane & 15, h = lane >> 4;
  _Float16* dst = out + (size_t)lid * 16;
  const float* row = W + (size_t)(nt * 16 + n) * K + kt * 32;
#pragma unroll
  for (int i = 0; i < 8; ++i) {
    const int k = 2 * i + 16 * h;
    dst[2 * i]     = (_Float16)row[k];
    dst[2 * i + 1] = (_Float16)row[k + 1];
  }
}

// From B-natural layout W[k][n] (ld=N).
__global__ void k_pack_w(const float* __restrict__ W, int K, int N,
                         _Float16* __restrict__ out) {
  const int lid = blockIdx.x * blockDim.x + threadIdx.x;
  const int ntn = N >> 4, ntk = K >> 5;
  if (lid >= ntk * ntn * 32) return;
  const int lane = lid & 31, tile = lid >> 5;
  const int nt = tile % ntn, kt = tile / ntn;
  const int n = lane & 15, h = lane >> 4;
  _Float16* dst = out + (size_t)lid * 16;
#pragma unroll
  for (int i = 0; i < 8; ++i) {
    const int k = kt * 32 + 2 * i + 16 * h;
    dst[2 * i]     = (_Float16)W[(size_t)k * N + nt * 16 + n];
    dst[2 * i + 1] = (_Float16)W[(size_t)(k + 1) * N + nt * 16 + n];
  }
}

// ---------------- 1) patch embed: Conv3d(4^3,s=4) as GEMM + BN + LIF ----------------
// out tokens in window-major order [win][tok][64], spikes as f16
__global__ void __launch_bounds__(256)
k_patch_embed(const float* __restrict__ x,        // [B,4,128,128,128] (one t)
              const _Float16* __restrict__ pe_pk, // packed [8x4 tiles]
              const float* __restrict__ pe_b,
              const float* __restrict__ pe_g,
              const float* __restrict__ pe_be,
              float* __restrict__ v_pe,           // [NTOK*64] membrane
              _Float16* __restrict__ spk)         // [NTOK*64]
{
  extern __shared__ char dynsmem[];
  _Float16* As = (_Float16*)dynsmem;              // [64][256]
  const int tid = threadIdx.x;
  const int g0  = blockIdx.x * 64;

  __builtin_prefetch(pe_pk + (size_t)tid * 64, 0, 3);

  // gather 64 tokens x 256 patch elements, fp32 -> f16
  for (int idx = tid; idx < 64 * 256; idx += 256) {
    const int m = idx >> 8, kk = idx & 255;
    int b, dz, dy, dx;
    tok_decode(g0 + m, b, dz, dy, dx);
    const int c  = kk >> 6;
    const int z  = dz * 4 + ((kk >> 4) & 3);
    const int y  = dy * 4 + ((kk >> 2) & 3);
    const int xx = dx * 4 + (kk & 3);
    const size_t src = (((size_t)(b * 4 + c) * 128 + z) * 128 + y) * 128 + xx;
    As[idx] = (_Float16)x[src];
  }
  __syncthreads();

  const int wave = tid >> 5, lane = tid & 31;
  const int nl = lane & 15, mh = lane >> 4;
  for (int tix = wave * 2; tix < wave * 2 + 2; ++tix) {   // 4x4 tiles, 2 per wave
    const int mt = tix >> 2, nt = tix & 3;
    v8f acc = {};
#pragma unroll
    for (int kk = 0; kk < 8; ++kk) {
      v16h a = frag_a_f16(&As[(mt * 16) * 256 + kk * 32], 256);
      v16h b = frag_b_pack(pe_pk, 4, kk, nt);
      acc = wmma_f16(a, b, acc);
    }
#pragma unroll
    for (int r = 0; r < 8; ++r) {
      const int ch = nt * 16 + nl;
      const int g  = g0 + mt * 16 + r + 8 * mh;
      float v = acc[r] + pe_b[ch];
      v = v * (pe_g[ch] * BN_SCALE) + pe_be[ch];
      const size_t o = (size_t)g * 64 + ch;
      spk[o] = (_Float16)lif_fire(&v_pe[o], v);
    }
  }
}

// ---------------- 2) windowed MHA + residual + LN + LIF (one block per window) ----------------
#define ATTN_SMEM (24576 + 16384 + 8192 + 8192 + 16384 + 512)
__global__ void __launch_bounds__(256)
k_win_attn(const _Float16* __restrict__ xin,     // [NWIN][64][64] spikes
           const _Float16* __restrict__ inw_pk,  // packed [2x12 tiles]
           const float* __restrict__ inb,        // [192]
           const _Float16* __restrict__ outw_pk, // packed [2x4 tiles]
           const float* __restrict__ outb,       // [64]
           const float* __restrict__ lng,
           const float* __restrict__ lnb,
           float* __restrict__ vmem,             // [NWIN*4096]
           _Float16* __restrict__ spk_out)
{
  extern __shared__ char dynsmem[];
  _Float16* QKV  = (_Float16*)dynsmem;                 // [64][192]
  float*    Sc   = (float*)   (dynsmem + 24576);       // [64][64]
  _Float16* Pf   = (_Float16*)(dynsmem + 40960);       // [64][64]
  _Float16* Of   = (_Float16*)(dynsmem + 49152);       // [64][64]
  float*    Ys   = (float*)   (dynsmem + 57344);       // [64][64]
  float*    stat = (float*)   (dynsmem + 73728);       // [128] mean|rstd

  const int tid = threadIdx.x, wave = tid >> 5, lane = tid & 31;
  const int nl = lane & 15, mh = lane >> 4;
  const _Float16* X = xin + (size_t)blockIdx.x * 4096;

  // ---- QKV = X @ inw^T + inb : M=64, N=192, K=64 ----
  for (int tix = wave; tix < 48; tix += 8) {
    const int mt = tix / 12, nt = tix % 12;
    v8f acc = {};
#pragma unroll
    for (int kk = 0; kk < 2; ++kk) {
      v16h a = frag_a_f16(X + (mt * 16) * 64 + kk * 32, 64);
      v16h b = frag_b_pack(inw_pk, 12, kk, nt);
      acc = wmma_f16(a, b, acc);
    }
#pragma unroll
    for (int r = 0; r < 8; ++r) {
      const int m = mt * 16 + r + 8 * mh, ch = nt * 16 + nl;
      QKV[m * 192 + ch] = (_Float16)(acc[r] + inb[ch]);
    }
  }
  __syncthreads();

  // ---- per head: scores (K=16 zero-padded), softmax, A@V ----
  for (int h = 0; h < 4; ++h) {
    for (int tix = wave; tix < 16; tix += 8) {
      const int mt = tix >> 2, nt = tix & 3;
      v16h a = frag_a_f16_klim(&QKV[(mt * 16) * 192 + h * 16], 192, 16);
      v16h b = frag_b_f16T_klim(&QKV[(nt * 16) * 192 + 64 + h * 16], 192, 16);
      v8f acc = {};
      acc = wmma_f16(a, b, acc);
#pragma unroll
      for (int r = 0; r < 8; ++r)
        Sc[(mt * 16 + r + 8 * mh) * 64 + nt * 16 + nl] = acc[r] * 0.25f; // dh^-0.5
    }
    __syncthreads();
    if (tid < 64) {                      // softmax over 64 keys, one row per thread
      float mx = -1e30f;
      for (int c = 0; c < 64; ++c) mx = fmaxf(mx, Sc[tid * 64 + c]);
      float sum = 0.0f;
      for (int c = 0; c < 64; ++c) {
        const float e = __expf(Sc[tid * 64 + c] - mx);
        Sc[tid * 64 + c] = e; sum += e;
      }
      const float inv = 1.0f / sum;
      for (int c = 0; c < 64; ++c) Pf[tid * 64 + c] = (_Float16)(Sc[tid * 64 + c] * inv);
    }
    __syncthreads();
    if (wave < 4) {                      // O_h = P @ V_h : M=64, N=16, K=64
      const int mt = wave;
      v8f acc = {};
#pragma unroll
      for (int kk = 0; kk < 2; ++kk) {
        v16h a = frag_a_f16(&Pf[(mt * 16) * 64 + kk * 32], 64);
        v16h b = frag_b_f16(&QKV[(kk * 32) * 192 + 128 + h * 16], 192);
        acc = wmma_f16(a, b, acc);
      }
#pragma unroll
      for (int r = 0; r < 8; ++r)
        Of[(mt * 16 + r + 8 * mh) * 64 + h * 16 + nl] = (_Float16)acc[r];
    }
    __syncthreads();
  }

  // ---- out proj + residual ----
  for (int tix = wave * 2; tix < wave * 2 + 2; ++tix) {
    const int mt = tix >> 2, nt = tix & 3;
    v8f acc = {};
#pragma unroll
    for (int kk = 0; kk < 2; ++kk) {
      v16h a = frag_a_f16(&Of[(mt * 16) * 64 + kk * 32], 64);
      v16h b = frag_b_pack(outw_pk, 4, kk, nt);
      acc = wmma_f16(a, b, acc);
    }
#pragma unroll
    for (int r = 0; r < 8; ++r) {
      const int m = mt * 16 + r + 8 * mh, ch = nt * 16 + nl;
      Ys[m * 64 + ch] = acc[r] + outb[ch] + (float)X[m * 64 + ch];
    }
  }
  __syncthreads();

  // ---- LayerNorm stats ----
  if (tid < 64) {
    float s = 0.0f;
    for (int c = 0; c < 64; ++c) s += Ys[tid * 64 + c];
    const float mean = s * (1.0f / 64.0f);
    float vs = 0.0f;
    for (int c = 0; c < 64; ++c) { const float d = Ys[tid * 64 + c] - mean; vs += d * d; }
    stat[tid]      = mean;
    stat[64 + tid] = rsqrtf(vs * (1.0f / 64.0f) + LN_EPSF);
  }
  __syncthreads();

  // ---- LN + LIF ----
  for (int idx = tid; idx < 4096; idx += 256) {
    const int m = idx >> 6, ch = idx & 63;
    const float y = (Ys[idx] - stat[m]) * stat[64 + m] * lng[ch] + lnb[ch];
    const size_t o = (size_t)blockIdx.x * 4096 + idx;
    spk_out[o] = (_Float16)lif_fire(&vmem[o], y);
  }
}

// ---------------- 3) 1x1 channel-mix GEMM + BN + LIF ----------------
__global__ void __launch_bounds__(256)
k_mlp_bn_lif(const _Float16* __restrict__ xin,  // [NTOK][64] spikes
             const _Float16* __restrict__ w_pk, // packed [2x4 tiles]
             const float* __restrict__ bias,
             const float* __restrict__ g,
             const float* __restrict__ be,
             float* __restrict__ vmem,
             _Float16* __restrict__ spk)
{
  const int tid = threadIdx.x, wave = tid >> 5, lane = tid & 31;
  const int nl = lane & 15, mh = lane >> 4;
  const int g0 = blockIdx.x * 64;
  for (int tix = wave * 2; tix < wave * 2 + 2; ++tix) {
    const int mt = tix >> 2, nt = tix & 3;
    v8f acc = {};
#pragma unroll
    for (int kk = 0; kk < 2; ++kk) {
      v16h a = frag_a_f16(xin + (size_t)(g0 + mt * 16) * 64 + kk * 32, 64);
      v16h b = frag_b_pack(w_pk, 4, kk, nt);
      acc = wmma_f16(a, b, acc);
    }
#pragma unroll
    for (int r = 0; r < 8; ++r) {
      const int ch = nt * 16 + nl;
      const size_t o = (size_t)(g0 + mt * 16 + r + 8 * mh) * 64 + ch;
      float v = acc[r] + bias[ch];
      v = v * (g[ch] * BN_SCALE) + be[ch];
      spk[o] = (_Float16)lif_fire(&vmem[o], v);
    }
  }
}

// ---------------- 4) ConvTranspose3d(2^3,s=2) on concat(x2,x1) + BN + LIF ----------------
// GEMM: M = tokens (64/block), N = 256 (32 out ch x 8 subpos), K = 128
__global__ void __launch_bounds__(256)
k_decoder(const _Float16* __restrict__ x2, const _Float16* __restrict__ x1,
          const _Float16* __restrict__ dec_pk, // packed [4x16 tiles]
          const float* __restrict__ dec_b,     // [32]
          const float* __restrict__ dec_g, const float* __restrict__ dec_be,
          float* __restrict__ vmem,            // [DEC_POS*32], voxel-major
          _Float16* __restrict__ spk)          // same layout
{
  const int tid = threadIdx.x, wave = tid >> 5, lane = tid & 31;
  const int nl = lane & 15, mh = lane >> 4;
  const int g0 = blockIdx.x * 64;

  __builtin_prefetch(dec_pk + (size_t)tid * 128, 0, 3);

  for (int tix = wave; tix < 64; tix += 8) {    // 4 x 16 tiles, 8 per wave
    const int mt = tix >> 4, nt = tix & 15;
    v8f acc = {};
#pragma unroll
    for (int kk = 0; kk < 4; ++kk) {            // concat: K 0..63 = x2, 64..127 = x1
      const _Float16* Ap = (kk < 2) ? (x2 + (size_t)(g0 + mt * 16) * 64 + kk * 32)
                                    : (x1 + (size_t)(g0 + mt * 16) * 64 + (kk - 2) * 32);
      v16h a = frag_a_f16(Ap, 64);
      v16h b = frag_b_pack(dec_pk, 16, kk, nt);
      acc = wmma_f16(a, b, acc);
    }
#pragma unroll
    for (int r = 0; r < 8; ++r) {
      const int g = g0 + mt * 16 + r + 8 * mh;
      const int n = nt * 16 + nl;
      const int o = n >> 3, sp = n & 7;
      int b, dz, dy, dx;
      tok_decode(g, b, dz, dy, dx);
      const int z = dz * 2 + ((sp >> 2) & 1);
      const int y = dy * 2 + ((sp >> 1) & 1);
      const int xx = dx * 2 + (sp & 1);
      const size_t vox = (size_t)b * 262144 + ((size_t)z * 64 + y) * 64 + xx;
      const size_t off = vox * 32 + o;
      float v = acc[r] + dec_b[o];
      v = v * (dec_g[o] * BN_SCALE) + dec_be[o];
      spk[off] = (_Float16)lif_fire(&vmem[off], v);
    }
  }
}

// ---------------- 5) readout 32->4 + temporal accumulate ----------------
__global__ void __launch_bounds__(256)
k_readout(const _Float16* __restrict__ spk,   // [DEC_POS][32]
          const float* __restrict__ ro_w,     // [4][32]
          const float* __restrict__ ro_b,
          float* __restrict__ out,            // [B][4][64^3]
          float tscale)
{
  const int p = blockIdx.x * blockDim.x + threadIdx.x;
  if (p >= DEC_POS) return;
  const int b = p >> 18, s = p & 262143;
  float acc[4] = {ro_b[0], ro_b[1], ro_b[2], ro_b[3]};
  const _Float16* sp = spk + (size_t)p * 32;
#pragma unroll
  for (int c = 0; c < 32; ++c) {
    const float v = (float)sp[c];
#pragma unroll
    for (int o = 0; o < 4; ++o) acc[o] += v * ro_w[o * 32 + c];
  }
#pragma unroll
  for (int o = 0; o < 4; ++o) {
    const size_t oi = (((size_t)b * 4 + o) << 18) + s;
    out[oi] += acc[o] * tscale;
  }
}

// ---------------- host launch ----------------
extern "C" void kernel_launch(void* const* d_in, const int* in_sizes, int n_in,
                              void* d_out, int out_size, void* d_ws, size_t ws_size,
                              hipStream_t stream) {
  (void)in_sizes; (void)n_in; (void)out_size; (void)ws_size;
  const float* x_seq = (const float*)d_in[0];
  const float* pe_w  = (const float*)d_in[1];
  const float* pe_b  = (const float*)d_in[2];
  const float* pe_g  = (const float*)d_in[3];
  const float* pe_be = (const float*)d_in[4];
  const float* blk[2][10];
  for (int bb = 0; bb < 2; ++bb)
    for (int i = 0; i < 10; ++i) blk[bb][i] = (const float*)d_in[5 + bb * 10 + i];
  // blk[b]: 0 inw, 1 inb, 2 outw, 3 outb, 4 lng, 5 lnb, 6 mlpw, 7 mlpb, 8 bng, 9 bnb
  const float* dec_w  = (const float*)d_in[25];
  const float* dec_b  = (const float*)d_in[26];
  const float* dec_g  = (const float*)d_in[27];
  const float* dec_be = (const float*)d_in[28];
  const float* ro_w   = (const float*)d_in[29];
  const float* ro_b   = (const float*)d_in[30];

  // workspace carve (bytes)
  char* ws = (char*)d_ws;
  const size_t MEMB = (size_t)NTOK * 64 * sizeof(float);       // 16 MB each
  float* v_pe  = (float*)(ws);
  float* v1a   = (float*)(ws + 1 * MEMB);
  float* v1m   = (float*)(ws + 2 * MEMB);
  float* v2a   = (float*)(ws + 3 * MEMB);
  float* v2m   = (float*)(ws + 4 * MEMB);
  float* v_dec = (float*)(ws + 5 * MEMB);                      // 64 MB
  char* sp0 = ws + 5 * MEMB + (size_t)DEC_POS * 32 * sizeof(float);
  const size_t SPK = (size_t)NTOK * 64 * sizeof(_Float16);     // 8 MB each
  _Float16* spk_pe  = (_Float16*)(sp0);
  _Float16* spk_ya  = (_Float16*)(sp0 + 1 * SPK);              // attn out (both blocks)
  _Float16* spk_x1  = (_Float16*)(sp0 + 2 * SPK);
  _Float16* spk_x2  = (_Float16*)(sp0 + 3 * SPK);
  _Float16* spk_dec = (_Float16*)(sp0 + 4 * SPK);              // 32 MB

  // packed-weight region (f16 WMMA-B fragments), 32B aligned
  char* pk = sp0 + 4 * SPK + (size_t)DEC_POS * 32 * sizeof(_Float16);
  _Float16* pe_pk    = (_Float16*)(pk);                        // 8x4 tiles  -> 32 KB
  _Float16* inw_pk0  = (_Float16*)(pk + 32768);                // 2x12 tiles -> 24 KB
  _Float16* inw_pk1  = (_Float16*)(pk + 32768 + 24576);
  _Float16* outw_pk0 = (_Float16*)(pk + 32768 + 2 * 24576);    // 2x4 tiles  -> 8 KB
  _Float16* outw_pk1 = (_Float16*)(pk + 32768 + 2 * 24576 + 8192);
  _Float16* mlpw_pk0 = (_Float16*)(pk + 32768 + 2 * 24576 + 2 * 8192);
  _Float16* mlpw_pk1 = (_Float16*)(pk + 32768 + 2 * 24576 + 3 * 8192);
  _Float16* dec_pk   = (_Float16*)(pk + 32768 + 2 * 24576 + 4 * 8192); // 4x16 -> 64 KB

  // zero membranes (contiguous prefix) and the accumulated output
  const long long membN = (long long)(5 * MEMB) / 4 + (long long)DEC_POS * 32;
  k_zero<<<2048, 256, 0, stream>>>((float*)d_ws, membN);
  k_zero<<<256, 256, 0, stream>>>((float*)d_out, (long long)2 * 4 * 262144);

  // pack weights once (f32 -> f16 fragment layout)
  k_pack_wT<<<4, 256, 0, stream>>>(pe_w, 256, 64, pe_pk);
  k_pack_wT<<<3, 256, 0, stream>>>(blk[0][0], 64, 192, inw_pk0);
  k_pack_wT<<<3, 256, 0, stream>>>(blk[1][0], 64, 192, inw_pk1);
  k_pack_wT<<<1, 256, 0, stream>>>(blk[0][2], 64, 64, outw_pk0);
  k_pack_wT<<<1, 256, 0, stream>>>(blk[1][2], 64, 64, outw_pk1);
  k_pack_wT<<<1, 256, 0, stream>>>(blk[0][6], 64, 64, mlpw_pk0);
  k_pack_wT<<<1, 256, 0, stream>>>(blk[1][6], 64, 64, mlpw_pk1);
  k_pack_w<<<8, 256, 0, stream>>>(dec_w, 128, 256, dec_pk);

  const size_t t_stride = (size_t)2 * 4 * 128 * 128 * 128;     // B*Cin*S^3
  for (int t = 0; t < 2; ++t) {
    const float* xt = x_seq + (size_t)t * t_stride;
    k_patch_embed<<<1024, 256, 64 * 256 * 2, stream>>>(xt, pe_pk, pe_b, pe_g, pe_be,
                                                       v_pe, spk_pe);
    k_win_attn<<<1024, 256, ATTN_SMEM, stream>>>(spk_pe, inw_pk0, blk[0][1],
                                                 outw_pk0, blk[0][3], blk[0][4],
                                                 blk[0][5], v1a, spk_ya);
    k_mlp_bn_lif<<<1024, 256, 0, stream>>>(spk_ya, mlpw_pk0, blk[0][7], blk[0][8],
                                           blk[0][9], v1m, spk_x1);
    k_win_attn<<<1024, 256, ATTN_SMEM, stream>>>(spk_x1, inw_pk1, blk[1][1],
                                                 outw_pk1, blk[1][3], blk[1][4],
                                                 blk[1][5], v2a, spk_ya);
    k_mlp_bn_lif<<<1024, 256, 0, stream>>>(spk_ya, mlpw_pk1, blk[1][7], blk[1][8],
                                           blk[1][9], v2m, spk_x2);
    k_decoder<<<1024, 256, 0, stream>>>(spk_x2, spk_x1, dec_pk, dec_b, dec_g, dec_be,
                                        v_dec, spk_dec);
    k_readout<<<DEC_POS / 256, 256, 0, stream>>>(spk_dec, ro_w, ro_b, (float*)d_out,
                                                 0.5f);
  }
}
